// proposal_extractor_65326452572716
// MI455X (gfx1250) — matrix operand order repaired and checked by reading, not compiled, not run on hardware
//
#include <hip/hip_runtime.h>
#include <hip/hip_bf16.h>

// ---------------- problem constants (match reference) ----------------
#define BATCH   8
#define NANCH   15360        // 96*160
#define NCLS    10
#define TOPK    100
#define GH0     96
#define GW0     160
#define IMGH    384.0f
#define IMGW    640.0f
#define FEATC   256
#define NLVL    4

// output layout (flat float32, reference return order)
#define O_ROI   ((size_t)0)                       // 4*8*100*256*49 = 40140800
#define O_BOX   ((size_t)40140800)                // 8*100*4 = 3200
#define O_CLS   ((size_t)40144000)                // 800
#define O_SCR   ((size_t)40144800)                // 800
#define O_QRY   ((size_t)40145600)                // 8*100*14 = 11200
#define O_KEEP  ((size_t)40156800)                // 800

typedef __attribute__((ext_vector_type(16))) _Float16 v16h;
typedef __attribute__((ext_vector_type(8)))  float    v8f;

union AV { v16h v; _Float16 h[16]; };
union CV { v8f  v; float    f[8];  };

// ---------------- K1: masked score per anchor ----------------
__global__ __launch_bounds__(256) void score_kernel(
    const float* __restrict__ clsl, const float* __restrict__ objl,
    const float* __restrict__ ctrl, const float* __restrict__ thr,
    float* __restrict__ msc)
{
    int i = blockIdx.x * blockDim.x + threadIdx.x;
    if (i >= BATCH * NANCH) return;
    const float* cl = clsl + (size_t)i * NCLS;
    float m = cl[0]; int am = 0;
#pragma unroll
    for (int j = 1; j < NCLS; ++j) { float v = cl[j]; if (v > m) { m = v; am = j; } }
    float s = 0.f;
#pragma unroll
    for (int j = 0; j < NCLS; ++j) s += expf(cl[j] - m);
    float prob = 1.f / s;                               // max of softmax
    float o  = 1.f / (1.f + expf(-objl[i]));
    float ce = 1.f / (1.f + expf(-ctrl[i]));
    float sc = sqrtf(prob * o * ce);
    msc[i] = (sc > thr[am]) ? sc : -1.0f;
}

// ---------------- K2: iterative top-100 per batch (32 waves) ----------------
__global__ __launch_bounds__(1024) void topk_kernel(
    float* __restrict__ msc, int* __restrict__ tidx, float* __restrict__ tval)
{
    int b = blockIdx.x;
    float* sc = msc + (size_t)b * NANCH;
    __shared__ float sv[1024];
    __shared__ int   si[1024];
    for (int it = 0; it < TOPK; ++it) {
        float bv = -1e30f; int bi = 0x7fffffff;
        for (int n = threadIdx.x; n < NANCH; n += 1024) {
            float v = sc[n];
            if (v > bv || (v == bv && n < bi)) { bv = v; bi = n; }
        }
        sv[threadIdx.x] = bv; si[threadIdx.x] = bi;
        __syncthreads();
        for (int off = 512; off > 0; off >>= 1) {
            if (threadIdx.x < off) {
                float v = sv[threadIdx.x + off]; int ii = si[threadIdx.x + off];
                if (v > sv[threadIdx.x] || (v == sv[threadIdx.x] && ii < si[threadIdx.x])) {
                    sv[threadIdx.x] = v; si[threadIdx.x] = ii;
                }
            }
            __syncthreads();
        }
        if (threadIdx.x == 0) {
            tidx[b * TOPK + it] = si[0];
            tval[b * TOPK + it] = sv[0];
            sc[si[0]] = -1e30f;                         // remove winner
        }
        __syncthreads();
    }
}

// ---------------- K3: decode + greedy NMS + small outputs ----------------
__global__ __launch_bounds__(128) void decode_nms_kernel(
    const float* __restrict__ deltas, const float* __restrict__ clsl,
    const float* __restrict__ objl,   const float* __restrict__ ctrl,
    const float* __restrict__ grid,   const float* __restrict__ dmean,
    const float* __restrict__ dstd,   const int* __restrict__ tidx,
    const float* __restrict__ tval,   float* __restrict__ out)
{
    int b = blockIdx.x, t = threadIdx.x;
    __shared__ float bx[TOPK][4];
    __shared__ float offc[TOPK];
    __shared__ int   keepA[TOPK];
    __shared__ float smv[TOPK][NCLS];
    __shared__ float objv[TOPK], ctrv[TOPK], scv[TOPK];
    __shared__ int   civ[TOPK];

    if (t < TOPK) {
        int n = tidx[b * TOPK + t];
        float ts = tval[b * TOPK + t];
        const float* cl = clsl + ((size_t)b * NANCH + n) * NCLS;
        float m = cl[0]; int am = 0;
#pragma unroll
        for (int j = 1; j < NCLS; ++j) { float v = cl[j]; if (v > m) { m = v; am = j; } }
        float s = 0.f; float e[NCLS];
#pragma unroll
        for (int j = 0; j < NCLS; ++j) { e[j] = expf(cl[j] - m); s += e[j]; }
#pragma unroll
        for (int j = 0; j < NCLS; ++j) smv[t][j] = e[j] / s;
        objv[t] = 1.f / (1.f + expf(-objl[(size_t)b * NANCH + n]));
        ctrv[t] = 1.f / (1.f + expf(-ctrl[(size_t)b * NANCH + n]));
        civ[t] = am; scv[t] = ts;
        const float* dd = deltas + ((size_t)b * NANCH + n) * 4;
        float d0 = dd[0] * dstd[0] + dmean[0];
        float d1 = dd[1] * dstd[1] + dmean[1];
        float d2 = dd[2] * dstd[2] + dmean[2];
        float d3 = dd[3] * dstd[3] + dmean[3];
        float gx = grid[(size_t)n * 2], gy = grid[(size_t)n * 2 + 1];
        bx[t][0] = fminf(fmaxf(gx - d0, 0.f), IMGW);
        bx[t][1] = fminf(fmaxf(gy - d1, 0.f), IMGH);
        bx[t][2] = fminf(fmaxf(gx + d2, 0.f), IMGW);
        bx[t][3] = fminf(fmaxf(gy + d3, 0.f), IMGH);
        keepA[t] = (ts > 0.f) ? 1 : 0;
        offc[t] = (float)am * (IMGW + 2.0f);            // max(img_w,img_h)+2 = 642
    }
    __syncthreads();

    for (int i = 0; i < TOPK; ++i) {
        __syncthreads();
        if (t < TOPK && t > i && keepA[i]) {
            float ax1 = bx[i][0] + offc[i], ay1 = bx[i][1] + offc[i];
            float ax2 = bx[i][2] + offc[i], ay2 = bx[i][3] + offc[i];
            float bx1 = bx[t][0] + offc[t], by1 = bx[t][1] + offc[t];
            float bx2 = bx[t][2] + offc[t], by2 = bx[t][3] + offc[t];
            float areaA = fmaxf(ax2 - ax1, 0.f) * fmaxf(ay2 - ay1, 0.f);
            float areaB = fmaxf(bx2 - bx1, 0.f) * fmaxf(by2 - by1, 0.f);
            float iw = fmaxf(fminf(ax2, bx2) - fmaxf(ax1, bx1), 0.f);
            float ih = fmaxf(fminf(ay2, by2) - fmaxf(ay1, by1), 0.f);
            float inter = iw * ih;
            float iou = inter / (areaA + areaB - inter + 1e-9f);
            if (iou > 0.5f) keepA[t] = 0;
        }
    }
    __syncthreads();

    if (t < TOPK) {
        float kf = (float)keepA[t];
        size_t base = (size_t)b * TOPK + t;
        float x1 = bx[t][0] * kf, y1 = bx[t][1] * kf;
        float x2 = bx[t][2] * kf, y2 = bx[t][3] * kf;
        out[O_BOX + base * 4 + 0] = x1;
        out[O_BOX + base * 4 + 1] = y1;
        out[O_BOX + base * 4 + 2] = x2;
        out[O_BOX + base * 4 + 3] = y2;
        out[O_CLS + base] = (float)civ[t];
        out[O_SCR + base] = scv[t] * kf;
        float* q = out + O_QRY + base * 14;
#pragma unroll
        for (int j = 0; j < NCLS; ++j) q[j] = smv[t][j] * kf;
        q[10] = objv[t] * kf;
        q[11] = ctrv[t] * kf;
        q[12] = (x2 - x1) * (1.0f / IMGW) * kf;
        q[13] = (y2 - y1) * (1.0f / IMGH) * kf;
        out[O_KEEP + base] = kf;
    }
}

// ---------------- K4: separable ROI-align via WMMA, zero LDS in hot loop ----
// GEMM1: Ut(rslot,px) = P(rslot x 32 colslots) * WxT(32 x 16px)   (2 WMMAs)
// GEMM2: out(px VGPRs, py lanes) = U(px x 32 rslots) * Wy(32 x 16py) (1 WMMA)
// GEMM1's D layout IS GEMM2's A layout per-lane (aU.h[i]=D0.f[i],
// aU.h[8+i]=D1.f[i]) -> no LDS transpose, no per-channel barriers.
__global__ __launch_bounds__(256) void roi_wmma_kernel(
    const float* __restrict__ f2, const float* __restrict__ f3,
    const float* __restrict__ f4, const float* __restrict__ f5,
    float* __restrict__ out)
{
    int blk = blockIdx.x;
    int lvl = blk / (BATCH * TOPK);
    int rem = blk % (BATCH * TOPK);
    int bb = rem / TOPK;
    int kk = rem % TOPK;
    const float* feat = (lvl == 0) ? f2 : (lvl == 1) ? f3 : (lvl == 2) ? f4 : f5;
    int fh = GH0 >> lvl, fw = GW0 >> lvl;

    float keep_f = out[O_KEEP + (size_t)bb * TOPK + kk];
    float* roi_out = out + O_ROI + (((size_t)lvl * BATCH + bb) * TOPK + kk) * (FEATC * 49);
    int tid = threadIdx.x;

    if (keep_f == 0.0f) {
        float4* ro4 = (float4*)roi_out;                  // 12544 floats = 3136 float4
        for (int i = tid; i < FEATC * 49 / 4; i += 256) ro4[i] = make_float4(0.f, 0.f, 0.f, 0.f);
        return;
    }

    __shared__ float    bxs[4];
    __shared__ _Float16 WxA[16][32];  // [px][col-slot] weights (B of GEMM1, via B-layout read)
    __shared__ _Float16 WyB[32][16];  // [row-slot][py]  weights (B of GEMM2)
    __shared__ int      rowIdx[32];   // row-slot -> feature row
    __shared__ int      colBase[16];  // sample -> base column of its (x0,x0+1) pair

    if (tid < 4) {
        float v = out[O_BOX + ((size_t)bb * TOPK + kk) * 4 + tid];
        float s = (tid & 1) ? ((float)fh / IMGH) : ((float)fw / IMGW);
        bxs[tid] = v * s;
    }
    for (int i = tid; i < 16 * 32; i += 256) ((_Float16*)WxA)[i] = (_Float16)0.f;
    for (int i = tid; i < 32 * 16; i += 256) ((_Float16*)WyB)[i] = (_Float16)0.f;
    if (tid < 32) rowIdx[tid] = 0;
    if (tid < 16) colBase[tid] = 0;
    __syncthreads();

    if (tid < 14) {
        int j = tid;
        float frac = (float)(j >> 1) + 0.25f + 0.5f * (float)(j & 1);
        float x1 = bxs[0], y1 = bxs[1], x2 = bxs[2], y2 = bxs[3];
        float bw = (x2 - x1) * (1.f / 7.f);
        float bh = (y2 - y1) * (1.f / 7.f);
        // x axis -> column pair + weights (border clamp folded into weights)
        float xs = x1 - 0.5f + bw * frac;
        float vx = (xs >= -1.f && xs <= (float)fw) ? 1.f : 0.f;
        float xc = fminf(fmaxf(xs, 0.f), (float)fw - 1.f);
        float xf = floorf(xc); float lx = xc - xf;
        int x0i = (int)xf;
        float wlo = 0.5f * vx * (1.f - lx);              // 0.5 = subsample avg
        float whi = 0.5f * vx * lx;
        int   xb = x0i; float w0 = wlo, w1 = whi;
        if (x0i > fw - 2) { xb = fw - 2; w0 = 0.f; w1 = wlo + whi; }  // lx==0 here
        colBase[j] = xb;
        int px = j >> 1;
        WxA[px][2 * j]     = (_Float16)w0;
        WxA[px][2 * j + 1] = (_Float16)w1;
        // y axis -> row slots + weights
        float ys = y1 - 0.5f + bh * frac;
        float vy = (ys >= -1.f && ys <= (float)fh) ? 1.f : 0.f;
        float yc = fminf(fmaxf(ys, 0.f), (float)fh - 1.f);
        float yf = floorf(yc); float ly = yc - yf;
        int y0i = (int)yf; int y1i = min(y0i + 1, fh - 1);
        rowIdx[2 * j] = y0i; rowIdx[2 * j + 1] = y1i;
        int py = j >> 1;
        WyB[2 * j][py]     = (_Float16)(0.5f * vy * (1.f - ly));
        WyB[2 * j + 1][py] = (_Float16)(0.5f * vy * ly);
    }
    __syncthreads();

    int lane = tid & 31;
    int wave = tid >> 5;
    int n  = lane & 15;       // A-matrix M row / B-matrix N column (0..15)
    int hi = lane >> 4;       // lane half

    // Hoisted B of GEMM1: WxT in 16-bit B-matrix 32x16 layout (k=colslot, n=px).
    AV bX;
#pragma unroll
    for (int v = 0; v < 8; ++v) {
        int kb = (v < 4) ? (hi * 8 + 2 * v) : (16 + hi * 8 + 2 * (v - 4));
        bX.h[2 * v]     = WxA[n][kb];
        bX.h[2 * v + 1] = WxA[n][kb + 1];
    }
    // Hoisted B of GEMM2: Wy in 16-bit B-matrix 32x16 layout (k=rowslot, n=py).
    AV bY;
#pragma unroll
    for (int v = 0; v < 8; ++v) {
        int kb = (v < 4) ? (hi * 8 + 2 * v) : (16 + hi * 8 + 2 * (v - 4));
        bY.h[2 * v]     = WyB[kb][n];
        bY.h[2 * v + 1] = WyB[kb + 1][n];
    }
    // Gather metadata for the A operands (patch rows).
    // A-layout k-pairs for lane half 'hi': v<4 -> k=hi*8+2v, v>=4 -> k=16+hi*8+2(v-4);
    // column-pair sample s = k/2.
    int cbs[8];
#pragma unroll
    for (int v = 0; v < 8; ++v) {
        int s = (v < 4) ? (hi * 4 + v) : (8 + hi * 4 + (v - 4));
        cbs[v] = colBase[s];
    }
    int rA0 = rowIdx[n];        // A0 row (row-slots 0..15): this lane's matrix row
    int rA1 = rowIdx[16 + n];   // A1 row (row-slots 16..31)

    const float* fbase = feat + (size_t)bb * FEATC * fh * fw;
    for (int t = 0; t < FEATC / 8; ++t) {
        int c = wave * (FEATC / 8) + t;
        const float* fc = fbase + (size_t)c * fh * fw;
        __builtin_prefetch(fc + (size_t)fh * fw, 0, 1);   // next channel -> global_prefetch_b8
        const float* rp0 = fc + (size_t)rA0 * fw;
        const float* rp1 = fc + (size_t)rA1 * fw;

        AV a0, a1;                                        // patch rows in A-layout
#pragma unroll
        for (int v = 0; v < 8; ++v) {
            int cb = cbs[v];
            float p00 = rp0[cb], p01 = rp0[cb + 1];       // adjacent -> b64
            float p10 = rp1[cb], p11 = rp1[cb + 1];       // adjacent -> b64
            a0.h[2 * v]     = (_Float16)p00;
            a0.h[2 * v + 1] = (_Float16)p01;
            a1.h[2 * v]     = (_Float16)p10;
            a1.h[2 * v + 1] = (_Float16)p11;
        }
        CV t0, t1; t0.v = (v8f){0.f}; t1.v = (v8f){0.f};
        t0.v = __builtin_amdgcn_wmma_f32_16x16x32_f16(false, a0.v, false, bX.v,
                                                      (short)0, t0.v, false, false);
        t1.v = __builtin_amdgcn_wmma_f32_16x16x32_f16(false, a1.v, false, bX.v,
                                                      (short)0, t1.v, false, false);
        // GEMM1 D layout == GEMM2 A layout, purely in-lane repack:
        AV aU;
#pragma unroll
        for (int i = 0; i < 8; ++i) {
            aU.h[i]     = (_Float16)t0.f[i];
            aU.h[8 + i] = (_Float16)t1.f[i];
        }
        CV d; d.v = (v8f){0.f};
        d.v = __builtin_amdgcn_wmma_f32_16x16x32_f16(false, aU.v, false, bY.v,
                                                     (short)0, d.v, false, false);
        // D element (m=px, n=py): lane l<7 holds py=l, VGPR j holds px=j.
        // Each lane writes 7 CONTIGUOUS floats -> wide stores.
        if (hi == 0 && n < 7) {
#pragma unroll
            for (int j = 0; j < 7; ++j)
                roi_out[(size_t)c * 49 + n * 7 + j] = d.f[j];
        }
    }
}

// ---------------- launcher ----------------
extern "C" void kernel_launch(void* const* d_in, const int* in_sizes, int n_in,
                              void* d_out, int out_size, void* d_ws, size_t ws_size,
                              hipStream_t stream) {
    const float* f2     = (const float*)d_in[0];
    const float* f3     = (const float*)d_in[1];
    const float* f4     = (const float*)d_in[2];
    const float* f5     = (const float*)d_in[3];
    const float* deltas = (const float*)d_in[4];
    const float* objl   = (const float*)d_in[5];
    const float* clsl   = (const float*)d_in[6];
    const float* ctrl   = (const float*)d_in[7];
    const float* grid   = (const float*)d_in[8];
    const float* dmean  = (const float*)d_in[9];
    const float* dstd   = (const float*)d_in[10];
    const float* thr    = (const float*)d_in[11];
    float* out = (float*)d_out;

    float* wsf  = (float*)d_ws;
    float* msc  = wsf;                                   // 8*15360 floats
    int*   tidx = (int*)(wsf + BATCH * NANCH);           // 800 ints
    float* tval = wsf + BATCH * NANCH + BATCH * TOPK;    // 800 floats

    int total = BATCH * NANCH;
    score_kernel<<<(total + 255) / 256, 256, 0, stream>>>(clsl, objl, ctrl, thr, msc);
    topk_kernel<<<BATCH, 1024, 0, stream>>>(msc, tidx, tval);
    decode_nms_kernel<<<BATCH, 128, 0, stream>>>(deltas, clsl, objl, ctrl, grid,
                                                 dmean, dstd, tidx, tval, out);
    roi_wmma_kernel<<<NLVL * BATCH * TOPK, 256, 0, stream>>>(f2, f3, f4, f5, out);
}
